// HypoNerf_31671088840755
// MI455X (gfx1250) — matrix operand order repaired and checked by reading, not compiled, not run on hardware
//
#include <hip/hip_runtime.h>
#include <hip/hip_bf16.h>

typedef __attribute__((ext_vector_type(16))) _Float16 v16h;
typedef __attribute__((ext_vector_type(8)))  float    v8f;
typedef __attribute__((ext_vector_type(4)))  unsigned int u32x4;
typedef __attribute__((ext_vector_type(2)))  float    f32x2;

union FragH { v16h h; u32x4 u[2]; };

#define HN_B        16
#define HN_N        16384
#define HN_H        256
#define BLOCK       128          // 4 wave32
#define ROWS        128          // points per workgroup (32 per wave)
#define ACT_STRIDE  264          // halfs per activation row (16B-aligned rows, conflict pad)
#define WT_STRIDE   40           // halfs per transposed-weight row (16B-aligned)
#define SMEM_HALFS  (2 * ROWS * ACT_STRIDE + HN_H * WT_STRIDE)
#define SMEM_BYTES  (SMEM_HALFS * 2)

// Load B fragment for column-tile nt from transposed weight slice Wt[n][k0..k0+31].
__device__ __forceinline__ void load_bfrag(FragH& bf, const _Float16* __restrict__ Wt,
                                           int nt, int lm, int lanehi)
{
    const int n  = (nt << 4) + lm;
    const int kb = lanehi * 8;
    bf.u[0] = *(const u32x4*)(Wt + n * WT_STRIDE + kb);
    bf.u[1] = *(const u32x4*)(Wt + n * WT_STRIDE + kb + 16);
}

// One ReLU GEMM layer: actOut[ROWS][H] = relu(actIn[ROWS][Kpad] * W[Kreal][H] + bias)
// W is f32 row-major [Kreal+1][H] in global (last row = bias). Kpad multiple of 32;
// actIn columns [Kreal, Kpad) must be zero. Each wave owns 32 rows (2 x 16-row tiles).
__device__ __forceinline__ void gemm_relu_layer(const _Float16* __restrict__ actIn,
                                                _Float16* __restrict__ actOut,
                                                const float* __restrict__ Wg,
                                                _Float16* __restrict__ Wt,
                                                int Kreal, int Kpad, int tid)
{
    const int lane   = tid & 31;
    const int wave   = tid >> 5;       // 0..3
    const int lanehi = lane >> 4;      // 0 or 1
    const int lm     = lane & 15;
    const int r0     = wave << 5;      // first of this wave's 32 rows

    v8f zero = {0.f, 0.f, 0.f, 0.f, 0.f, 0.f, 0.f, 0.f};
    v8f accLo[16], accHi[16];
#pragma unroll
    for (int i = 0; i < 16; ++i) { accLo[i] = zero; accHi[i] = zero; }

    const int nChunks = Kpad >> 5;
    for (int kc = 0; kc < nChunks; ++kc) {
        const int kbase = kc << 5;
        __syncthreads();                       // Wt free to overwrite
        // Cooperative load W[kbase..kbase+31][0..255] -> Wt[n][k] (f16, transposed).
        // Thread i of iteration handles (k = i>>7, n-pair = i&127): float2 reads coalesce.
#pragma unroll 4
        for (int i = tid; i < 32 * 128; i += BLOCK) {
            const int j  = i & 127;            // n pair index
            const int k  = i >> 7;             // 0..31
            const int kg = kbase + k;
            f32x2 w = {0.f, 0.f};
            if (kg < Kreal) w = *(const f32x2*)(Wg + kg * HN_H + 2 * j);
            Wt[(2 * j)     * WT_STRIDE + k] = (_Float16)w.x;
            Wt[(2 * j + 1) * WT_STRIDE + k] = (_Float16)w.y;
        }
        __syncthreads();                       // Wt ready

        // Two A fragments: rows r0..r0+15 and r0+16..r0+31, K slice kbase..kbase+31
        FragH a0, a1;
        const int k0 = kbase + lanehi * 8;
        a0.u[0] = *(const u32x4*)(actIn + (r0 + lm)      * ACT_STRIDE + k0);
        a0.u[1] = *(const u32x4*)(actIn + (r0 + lm)      * ACT_STRIDE + k0 + 16);
        a1.u[0] = *(const u32x4*)(actIn + (r0 + 16 + lm) * ACT_STRIDE + k0);
        a1.u[1] = *(const u32x4*)(actIn + (r0 + 16 + lm) * ACT_STRIDE + k0 + 16);

        // Double-buffered B fragments: each amortized over 2 WMMAs (both row tiles)
        FragH bf0, bf1;
        load_bfrag(bf0, Wt, 0, lm, lanehi);
#pragma unroll
        for (int nt = 0; nt < 16; nt += 2) {
            load_bfrag(bf1, Wt, nt + 1, lm, lanehi);
            accLo[nt] = __builtin_amdgcn_wmma_f32_16x16x32_f16(
                false, a0.h, false, bf0.h, (short)0, accLo[nt], false, false);
            accHi[nt] = __builtin_amdgcn_wmma_f32_16x16x32_f16(
                false, a1.h, false, bf0.h, (short)0, accHi[nt], false, false);
            if (nt + 2 < 16) load_bfrag(bf0, Wt, nt + 2, lm, lanehi);
            accLo[nt + 1] = __builtin_amdgcn_wmma_f32_16x16x32_f16(
                false, a0.h, false, bf1.h, (short)0, accLo[nt + 1], false, false);
            accHi[nt + 1] = __builtin_amdgcn_wmma_f32_16x16x32_f16(
                false, a1.h, false, bf1.h, (short)0, accHi[nt + 1], false, false);
        }
    }

    // bias + relu + store f16 (each wave writes only its own 32 rows)
    const float* __restrict__ bias = Wg + (size_t)Kreal * HN_H;
#pragma unroll
    for (int nt = 0; nt < 16; ++nt) {
        const int n  = (nt << 4) + lm;
        const float bv = bias[n];
        const int rb0 = r0 + lanehi * 8;
        const int rb1 = r0 + 16 + lanehi * 8;
#pragma unroll
        for (int r = 0; r < 8; ++r) {
            float v0 = accLo[nt][r] + bv;
            v0 = v0 > 0.0f ? v0 : 0.0f;
            actOut[(rb0 + r) * ACT_STRIDE + n] = (_Float16)v0;
            float v1 = accHi[nt][r] + bv;
            v1 = v1 > 0.0f ? v1 : 0.0f;
            actOut[(rb1 + r) * ACT_STRIDE + n] = (_Float16)v1;
        }
    }
}

__global__ __launch_bounds__(BLOCK) void hyponerf_kernel(
    const float* __restrict__ x,       // [B,N,3]
    const float* __restrict__ wb0,     // [B,121,256]
    const float* __restrict__ wb1,     // [B,257,256]
    const float* __restrict__ wb2,
    const float* __restrict__ wb3,
    const float* __restrict__ wb4,
    const float* __restrict__ wb_out,  // [B,257,1]
    const float* __restrict__ freqs,   // [60,3]
    float* __restrict__ out)           // [B,N]
{
    extern __shared__ _Float16 smem[];
    _Float16* actA = smem;                                // ROWS x ACT_STRIDE
    _Float16* actB = actA + ROWS * ACT_STRIDE;            // ROWS x ACT_STRIDE
    _Float16* Wt   = actB + ROWS * ACT_STRIDE;            // H x WT_STRIDE

    const int tid = threadIdx.x;
    const int bid = blockIdx.x;
    const int b   = bid >> 7;                 // 128 point-groups per batch
    const int pg  = bid & 127;
    const int p0  = pg * ROWS;

    // ---------- Phase 0: Gaussian positional encoding ----------
    // Stage scaled freqs + x tile in the (currently unused) Wt region.
    float* fshared = (float*)Wt;              // 180 floats
    float* xsh     = fshared + 192;           // 384 floats
    if (tid < 180 - 128) fshared[128 + tid] = freqs[128 + tid] * 20.0f;
    fshared[tid] = freqs[tid] * 20.0f;        // PE_DIM//2 = 20
    {
        const float* xb = x + ((size_t)b * HN_N + p0) * 3;
        for (int i = tid; i < ROWS * 3; i += BLOCK) xsh[i] = xb[i];
    }
    __syncthreads();

    for (int i = tid; i < ROWS * 60; i += BLOCK) {
        const int p = i / 60, f = i - p * 60;
        const float k = xsh[p * 3 + 0] * fshared[f * 3 + 0]
                      + xsh[p * 3 + 1] * fshared[f * 3 + 1]
                      + xsh[p * 3 + 2] * fshared[f * 3 + 2];
        actA[p * ACT_STRIDE + f]      = (_Float16)__sinf(k);
        actA[p * ACT_STRIDE + 60 + f] = (_Float16)__cosf(k);
    }
    for (int i = tid; i < ROWS * 8; i += BLOCK) {         // zero pad cols 120..127
        const int p = i >> 3, c = 120 + (i & 7);
        actA[p * ACT_STRIDE + c] = (_Float16)0.0f;
    }
    // gemm's first __syncthreads() covers PE->layer0 and fshared->Wt reuse.

    // ---------- 5 hidden layers (WMMA) ----------
    gemm_relu_layer(actA, actB, wb0 + (size_t)b * 121 * HN_H, Wt, 120, 128, tid);
    gemm_relu_layer(actB, actA, wb1 + (size_t)b * 257 * HN_H, Wt, 256, 256, tid);
    gemm_relu_layer(actA, actB, wb2 + (size_t)b * 257 * HN_H, Wt, 256, 256, tid);
    gemm_relu_layer(actB, actA, wb3 + (size_t)b * 257 * HN_H, Wt, 256, 256, tid);
    gemm_relu_layer(actA, actB, wb4 + (size_t)b * 257 * HN_H, Wt, 256, 256, tid);

    // ---------- Output layer: 256 -> 1 ----------
    __syncthreads();                                      // actB fully written by all waves
    const float* __restrict__ wo = wb_out + (size_t)b * 257;
    {
        float s = wo[256];
        const _Float16* hrow = actB + tid * ACT_STRIDE;
#pragma unroll 8
        for (int h = 0; h < HN_H; ++h) s += (float)hrow[h] * wo[h];
        out[(size_t)b * HN_N + p0 + tid] = s;
    }
}

extern "C" void kernel_launch(void* const* d_in, const int* in_sizes, int n_in,
                              void* d_out, int out_size, void* d_ws, size_t ws_size,
                              hipStream_t stream)
{
    (void)in_sizes; (void)n_in; (void)d_ws; (void)ws_size; (void)out_size;
    const float* x      = (const float*)d_in[0];
    const float* wb0    = (const float*)d_in[1];
    const float* wb1    = (const float*)d_in[2];
    const float* wb2    = (const float*)d_in[3];
    const float* wb3    = (const float*)d_in[4];
    const float* wb4    = (const float*)d_in[5];
    const float* wb_out = (const float*)d_in[6];
    const float* freqs  = (const float*)d_in[7];
    float* out = (float*)d_out;

    (void)hipFuncSetAttribute((const void*)hyponerf_kernel,
                              hipFuncAttributeMaxDynamicSharedMemorySize,
                              SMEM_BYTES);

    dim3 grid(HN_B * (HN_N / ROWS));   // 2048 workgroups
    dim3 block(BLOCK);                 // 4 wave32
    hyponerf_kernel<<<grid, block, SMEM_BYTES, stream>>>(
        x, wb0, wb1, wb2, wb3, wb4, wb_out, freqs, out);
}
